// Encoder_3530463118074
// MI455X (gfx1250) — matrix operand order repaired and checked by reading, not compiled, not run on hardware
//
#include <hip/hip_runtime.h>
#include <stdint.h>

// ---------- types ----------
typedef __attribute__((ext_vector_type(16))) __bf16 v16bf;
typedef __attribute__((ext_vector_type(8)))  __bf16 v8bf;
typedef __attribute__((ext_vector_type(8)))  float  v8f;
typedef unsigned int  u32x4 __attribute__((ext_vector_type(4)));
typedef int           i32x8 __attribute__((ext_vector_type(8)));
typedef int           i32x4 __attribute__((ext_vector_type(4)));
typedef unsigned short ushort_t;

#if defined(__has_builtin)
#  if __has_builtin(__builtin_amdgcn_tensor_load_to_lds)
#    define USE_TDM 1
#  endif
#endif
#ifndef USE_TDM
#  define USE_TDM 0
#endif

// ---------- scalar helpers ----------
__device__ __forceinline__ ushort_t f2bf(float f) {
  union { float f; unsigned u; } x; x.f = f;
  unsigned r = x.u + 0x7FFFu + ((x.u >> 16) & 1u);   // RNE
  return (ushort_t)(r >> 16);
}
__device__ __forceinline__ float sigm(float x) { return 1.0f / (1.0f + __expf(-x)); }

// ---------- WMMA helpers (CDNA5 16x16x32 bf16, f32 accum) ----------
__device__ __forceinline__ v8f wmma_bf16(v16bf a, v16bf b, v8f c) {
  return __builtin_amdgcn_wmma_f32_16x16x32_bf16(false, a, false, b, (short)0, c, false, false);
}

// A-matrix 16x32 bf16 fragment. lane = hi*16 + m.
// elem e in 0..7  -> K = k0 + hi*8 + e          (contiguous 8)
// elem e in 8..15 -> K = k0 + 16 + hi*8 + (e-8) (contiguous 8)
__device__ __forceinline__ v16bf load_fragA_row(const ushort_t* rowptr, int k0, int hi) {
  const ushort_t* p = rowptr + k0 + hi * 8;
  v8bf lo = *(const v8bf*)(p);
  v8bf hh = *(const v8bf*)(p + 16);
  return __builtin_shufflevector(lo, hh, 0,1,2,3,4,5,6,7,8,9,10,11,12,13,14,15);
}
__device__ __forceinline__ v16bf load_fragA(const ushort_t* base, int row, int ld, int k0, int hi) {
  return load_fragA_row(base + (size_t)row * ld, k0, hi);
}
// B-matrix 32x16 bf16 fragment from K-contiguous ("transposed") weights:
// lane = hi*16 + n;  elem e in 0..7 -> K = k0 + hi*8 + e; e in 8..15 -> K = k0 + 16 + hi*8 + (e-8)
__device__ __forceinline__ v16bf load_fragB(const ushort_t* baseT, int n, int ld, int k0, int hi) {
  const ushort_t* p = baseT + (size_t)n * ld + k0 + hi * 8;
  v8bf lo = *(const v8bf*)(p);
  v8bf hh = *(const v8bf*)(p + 16);
  return __builtin_shufflevector(lo, hh, 0,1,2,3,4,5,6,7,8,9,10,11,12,13,14,15);
}

// ---------- grid-wide sense-reversal barrier ----------
__device__ __forceinline__ void grid_sync(unsigned* cnt, unsigned* gen, unsigned nb) {
  __threadfence();
  __syncthreads();
  if (threadIdx.x == 0) {
    unsigned g = __hip_atomic_load(gen, __ATOMIC_ACQUIRE, __HIP_MEMORY_SCOPE_AGENT);
    unsigned a = __hip_atomic_fetch_add(cnt, 1u, __ATOMIC_ACQ_REL, __HIP_MEMORY_SCOPE_AGENT);
    if (a == nb - 1u) {
      __hip_atomic_store(cnt, 0u, __ATOMIC_RELAXED, __HIP_MEMORY_SCOPE_AGENT);
      __hip_atomic_fetch_add(gen, 1u, __ATOMIC_RELEASE, __HIP_MEMORY_SCOPE_AGENT);
    } else {
      while (__hip_atomic_load(gen, __ATOMIC_ACQUIRE, __HIP_MEMORY_SCOPE_AGENT) == g)
        __builtin_amdgcn_s_sleep(2);
    }
  }
  __syncthreads();
}

// ---------- small utility kernels ----------
__global__ void k_cast_bf16(const float* __restrict__ src, ushort_t* __restrict__ dst, size_t n) {
  for (size_t i = (size_t)blockIdx.x * blockDim.x + threadIdx.x; i < n;
       i += (size_t)gridDim.x * blockDim.x)
    dst[i] = f2bf(src[i]);
}

// dst[n*K+k] = bf16(src[k*src_ld + n])   (K-contiguous transposed weights)
__global__ void k_transpose_cast(const float* __restrict__ src, ushort_t* __restrict__ dst,
                                 int K, int N, int src_ld) {
  size_t total = (size_t)K * N;
  for (size_t id = (size_t)blockIdx.x * blockDim.x + threadIdx.x; id < total;
       id += (size_t)gridDim.x * blockDim.x) {
    size_t n = id / (size_t)K, k = id - n * (size_t)K;
    dst[n * (size_t)K + k] = f2bf(src[k * (size_t)src_ld + n]);
  }
}

__global__ void k_init_state(const float* __restrict__ state, float* __restrict__ c,
                             ushort_t* __restrict__ hbf) {
  int i = blockIdx.x * blockDim.x + threadIdx.x;
  if (i < 16 * 2048) c[i] = state[i];                       // state[0] : (16,2048)
  if (i < 16 * 1024) {
    int m = i >> 10, p = i & 1023;
    hbf[i] = f2bf(state[16 * 2048 + m * 2048 + p]);         // state[1][:, :1024]
  }
}

__global__ void k_zero_bar(unsigned* bar) { bar[0] = 0u; bar[1] = 0u; }

__global__ void k_gather_h(const float* __restrict__ hs, float* __restrict__ dsth, int Tcur) {
  int i = blockIdx.x * blockDim.x + threadIdx.x;          // 16*1024
  if (i < 16 * 1024) {
    int m = i >> 10, p = i & 1023;
    dsth[i] = hs[((size_t)m * Tcur + (Tcur - 1)) * 1024 + p];
  }
}

// ---------- xg = x @ Wx + bias   (M x 1024) x (1024 x 8192) ----------
// block: 256 thr (8 waves). WG tile = 16 rows x 128 cols. A slab staged to LDS via TDM.
__global__ void k_xg_gemm(const ushort_t* __restrict__ xbf, const ushort_t* __restrict__ wxT,
                          const float* __restrict__ bias, float* __restrict__ xg, int Mtot) {
  extern __shared__ ushort_t sA[];                         // 16 x 1024 bf16 = 32 KB
  const int tid = threadIdx.x, wid = tid >> 5, lane = tid & 31;
  const int hi = lane >> 4, ln = lane & 15;
  const int m0 = blockIdx.y * 16;

#if USE_TDM
  if (wid == 0) {  // one wave issues the tensor DMA (uniform branch)
    unsigned long long ga = (unsigned long long)(const void*)(xbf + (size_t)m0 * 1024);
    u32x4 g0 = { 1u,                                    // count=1 (valid descriptor)
                 0u,                                    // lds_addr = 0 (dynamic LDS base)
                 (unsigned)ga,
                 (unsigned)((ga >> 32) & 0x01FFFFFFull) | 0x80000000u };  // type=2
    i32x8 g1 = { 0x00010000,                            // data_size=1 (2 bytes)
                 0x04000000,                            // tensor_dim0=1024 (low16<<16)
                 0x00100000,                            // dim0 hi16 | tensor_dim1=16
                 0x04000000,                            // tensor_dim1 hi | tile_dim0=1024
                 16,                                    // tile_dim1=16
                 1024,                                  // tensor_dim0_stride
                 0, 0 };
    i32x4 z4 = { 0, 0, 0, 0 };
#if defined(__clang_major__) && (__clang_major__ >= 23)
    i32x8 z8 = { 0, 0, 0, 0, 0, 0, 0, 0 };
    __builtin_amdgcn_tensor_load_to_lds(g0, g1, z4, z4, z8, 0);
#else
    __builtin_amdgcn_tensor_load_to_lds(g0, g1, z4, z4, 0);
#endif
    __builtin_amdgcn_s_wait_tensorcnt(0);
  }
  __syncthreads();
#else
  {
    const uint4* src = (const uint4*)(xbf + (size_t)m0 * 1024);
    uint4* dst = (uint4*)sA;
    for (int i = tid; i < 2048; i += 256) dst[i] = src[i];
    __syncthreads();
  }
#endif

  const int n = blockIdx.x * 128 + wid * 16 + ln;
  v8f acc; { float bb = bias[n]; for (int r = 0; r < 8; ++r) acc[r] = bb; }
  for (int k0 = 0; k0 < 1024; k0 += 32) {
    v16bf a = load_fragA(sA, ln, 1024, k0, hi);          // ds_load
    v16bf b = load_fragB(wxT, n, 1024, k0, hi);
    acc = wmma_bf16(a, b, acc);
  }
  for (int r = 0; r < 8; ++r)
    xg[(size_t)(m0 + hi * 8 + r) * 8192 + n] = acc[r];
  (void)Mtot;
}

// ---------- persistent recurrent scan ----------
// grid = 8 blocks x 512 thr (16 waves) = 128 waves. Phase1: 128 d-tiles (4 gates each).
// Phase2: 64 proj n-tiles. Two grid barriers per timestep.
__global__ __launch_bounds__(512)
void k_lstm_scan(const float* __restrict__ xg, const ushort_t* __restrict__ whT,
                 const ushort_t* __restrict__ pjT, float* __restrict__ c,
                 ushort_t* __restrict__ hbf, ushort_t* __restrict__ mbf,
                 float* __restrict__ hs, int Tcur, unsigned* bar) {
  const int tid = threadIdx.x, wid = tid >> 5, lane = tid & 31;
  const int hi = lane >> 4, ln = lane & 15;
  const int gw = blockIdx.x * 16 + wid;                  // 0..127
  unsigned* cnt = bar; unsigned* gen = bar + 1;
  const unsigned nb = gridDim.x;

  for (int t = 0; t < Tcur; ++t) {
    { // ---- gates: g = h @ Wh (+ xg), per-wave d-tile of 16 ----
      const int d0 = gw * 16;
      const int d  = d0 + ln;
      v8f acc[4];
      for (int g = 0; g < 4; ++g) for (int r = 0; r < 8; ++r) acc[g][r] = 0.0f;
      for (int k0 = 0; k0 < 1024; k0 += 32) {
        v16bf a = load_fragA(hbf, ln, 1024, k0, hi);
        for (int g = 0; g < 4; ++g) {
          v16bf b = load_fragB(whT, g * 2048 + d, 1024, k0, hi);
          acc[g] = wmma_bf16(a, b, acc[g]);
        }
      }
      for (int r = 0; r < 8; ++r) {
        const int m = hi * 8 + r;
        const float* xr = xg + ((size_t)m * Tcur + t) * 8192;
        float gi = acc[0][r] + xr[d];
        float gj = acc[1][r] + xr[2048 + d];
        float gf = acc[2][r] + xr[4096 + d];
        float go = acc[3][r] + xr[6144 + d];
        float co = c[m * 2048 + d];
        float cn = sigm(gf + 1.0f) * co + sigm(gi) * tanhf(gj);
        c[m * 2048 + d] = cn;
        mbf[m * 2048 + d] = f2bf(sigm(go) * tanhf(cn));
      }
    }
    grid_sync(cnt, gen, nb);
    if (gw < 64) { // ---- projection: h = m @ proj ----
      const int n = gw * 16 + ln;
      v8f acc; for (int r = 0; r < 8; ++r) acc[r] = 0.0f;
      for (int k0 = 0; k0 < 2048; k0 += 32) {
        v16bf a = load_fragA(mbf, ln, 2048, k0, hi);
        v16bf b = load_fragB(pjT, n, 2048, k0, hi);
        acc = wmma_bf16(a, b, acc);
      }
      for (int r = 0; r < 8; ++r) {
        const int m = hi * 8 + r;
        float hv = acc[r];
        hs[((size_t)m * Tcur + t) * 1024 + n] = hv;
        hbf[m * 1024 + n] = f2bf(hv);
      }
    }
    grid_sync(cnt, gen, nb);
  }
}

// ---------- LayerNorm over last dim (P=1024), fused f32 + bf16 outputs ----------
__global__ void k_layernorm(const float* __restrict__ x, float* __restrict__ yf,
                            ushort_t* __restrict__ ybf, const float* __restrict__ g,
                            const float* __restrict__ b) {
  __shared__ float red[128];
  const int row = blockIdx.x, tid = threadIdx.x;
  const float* xr = x + (size_t)row * 1024;
  float s = 0.0f;
  for (int p = tid; p < 1024; p += 128) s += xr[p];
  red[tid] = s; __syncthreads();
  for (int o = 64; o > 0; o >>= 1) { if (tid < o) red[tid] += red[tid + o]; __syncthreads(); }
  float mean = red[0] * (1.0f / 1024.0f); __syncthreads();
  float v = 0.0f;
  for (int p = tid; p < 1024; p += 128) { float d = xr[p] - mean; v += d * d; }
  red[tid] = v; __syncthreads();
  for (int o = 64; o > 0; o >>= 1) { if (tid < o) red[tid] += red[tid + o]; __syncthreads(); }
  float rs = rsqrtf(red[0] * (1.0f / 1024.0f) + 1.0e-3f);
  for (int p = tid; p < 1024; p += 128) {
    float o = (xr[p] - mean) * rs * g[p] + b[p];
    yf[(size_t)row * 1024 + p] = o;
    ybf[(size_t)row * 1024 + p] = f2bf(o);
  }
}

// ---------- width-2 valid conv as 2-tap WMMA GEMM ----------
// in: (16,512,1024) bf16  ->  out: (16,511,1024) bf16. grid (8, 511), block 256.
__global__ void k_conv(const ushort_t* __restrict__ inbf, const ushort_t* __restrict__ cvT,
                       const float* __restrict__ convb, ushort_t* __restrict__ outbf) {
  const int tid = threadIdx.x, wid = tid >> 5, lane = tid & 31;
  const int hi = lane >> 4, ln = lane & 15;
  const int n = blockIdx.x * 128 + wid * 16 + ln;
  const int m0 = blockIdx.y * 16;
  v8f acc; { float bb = convb[n]; for (int r = 0; r < 8; ++r) acc[r] = bb; }
  for (int w = 0; w < 2; ++w) {
    const ushort_t* wt = cvT + (size_t)w * 1024 * 1024;
    const int mo = m0 + ln;
    const int bI = mo / 511, tt = mo - bI * 511;
    const ushort_t* arow = inbf + (size_t)(bI * 512 + tt + w) * 1024;
    for (int k0 = 0; k0 < 1024; k0 += 32) {
      v16bf a = load_fragA_row(arow, k0, hi);
      v16bf b = load_fragB(wt, n, 1024, k0, hi);
      acc = wmma_bf16(a, b, acc);
    }
  }
  for (int r = 0; r < 8; ++r)
    outbf[(size_t)(m0 + hi * 8 + r) * 1024 + n] = f2bf(acc[r]);
}

// ---------- host ----------
extern "C" void kernel_launch(void* const* d_in, const int* in_sizes, int n_in,
                              void* d_out, int out_size, void* d_ws, size_t ws_size,
                              hipStream_t stream) {
  const float* inputs  = (const float*)d_in[0];   // (16,512,1024)
  const float* state   = (const float*)d_in[1];   // (2,16,2048)
  const float* kernels = (const float*)d_in[2];   // (4,2048,8192)
  const float* biases  = (const float*)d_in[3];   // (4,8192)
  const float* projs   = (const float*)d_in[4];   // (4,2048,1024)
  const float* ln_g    = (const float*)d_in[5];   // (1024)
  const float* ln_b    = (const float*)d_in[6];   // (1024)
  const float* conv_w  = (const float*)d_in[7];   // (2,1024,1024)
  const float* conv_b  = (const float*)d_in[8];   // (1024)
  float* out = (float*)d_out;
  (void)in_sizes; (void)n_in; (void)out_size; (void)ws_size;

  char* ws = (char*)d_ws;
  size_t off = 0;
  auto carve = [&](size_t bytes) { char* p = ws + off; off += (bytes + 255) & ~(size_t)255; return p; };
  ushort_t* wxT  = (ushort_t*)carve((size_t)8192 * 1024 * 2);
  ushort_t* whT  = (ushort_t*)carve((size_t)8192 * 1024 * 2);
  ushort_t* pjT  = (ushort_t*)carve((size_t)1024 * 2048 * 2);
  ushort_t* cvT  = (ushort_t*)carve((size_t)2 * 1024 * 1024 * 2);
  ushort_t* X1   = (ushort_t*)carve((size_t)8192 * 1024 * 2);
  ushort_t* X2   = (ushort_t*)carve((size_t)8192 * 1024 * 2);
  float*    xg   = (float*)   carve((size_t)8192 * 8192 * 4);
  float*    hsb  = (float*)   carve((size_t)8192 * 1024 * 4);
  float*    cbuf = (float*)   carve((size_t)16 * 2048 * 4);
  ushort_t* hbf  = (ushort_t*)carve((size_t)16 * 1024 * 2);
  ushort_t* mbf  = (ushort_t*)carve((size_t)16 * 2048 * 2);
  unsigned* bar  = (unsigned*)carve(256);

  // one-time prep
  k_cast_bf16<<<2048, 256, 0, stream>>>(inputs, X1, (size_t)16 * 512 * 1024);
  k_transpose_cast<<<2048, 256, 0, stream>>>(conv_w,                cvT,                1024, 1024, 1024);
  k_transpose_cast<<<2048, 256, 0, stream>>>(conv_w + 1024 * 1024,  cvT + 1024 * 1024,  1024, 1024, 1024);
  k_init_state<<<(16 * 2048 + 255) / 256, 256, 0, stream>>>(state, cbuf, hbf);

  ushort_t* cur = X1;   // current layer input (bf16)
  ushort_t* oth = X2;
  int Tin = 512;
  for (int i = 0; i < 4; ++i) {
    const int Mtot = 16 * Tin;
    const float* Ki = kernels + (size_t)i * 2048 * 8192;
    k_transpose_cast<<<4096, 256, 0, stream>>>(Ki,                 wxT, 1024, 8192, 8192);
    k_transpose_cast<<<4096, 256, 0, stream>>>(Ki + 1024 * 8192,   whT, 1024, 8192, 8192);
    k_transpose_cast<<<2048, 256, 0, stream>>>(projs + (size_t)i * 2048 * 1024, pjT, 2048, 1024, 1024);

    k_xg_gemm<<<dim3(64, Mtot / 16), 256, 32768, stream>>>(cur, wxT, biases + (size_t)i * 8192, xg, Mtot);

    k_zero_bar<<<1, 1, 0, stream>>>(bar);
    k_lstm_scan<<<8, 512, 0, stream>>>(xg, whT, pjT, cbuf, hbf, mbf, hsb, Tin, bar);

    if (i == 3)   // final carry h = hs[:, -1, :]  (pre-LN)
      k_gather_h<<<(16 * 1024 + 255) / 256, 256, 0, stream>>>(hsb, out + 8404992, Tin);

    float* lnf = (i == 3) ? out : hsb;          // last layer LN writes d_out directly
    k_layernorm<<<Mtot, 128, 0, stream>>>(hsb, lnf, oth, ln_g, ln_b);

    if (i == 1) {             // conv reduces T 512 -> 511; reads LN-bf16, writes into `cur`
      k_conv<<<dim3(8, 511), 256, 0, stream>>>(oth, cvT, conv_b, cur);
      Tin = 511;              // `cur` already holds next layer input
    } else {
      ushort_t* t2 = cur; cur = oth; oth = t2;
    }
  }

  // final cell state c -> d_out tail
  hipMemcpyAsync(out + 8372224, cbuf, (size_t)16 * 2048 * 4, hipMemcpyDeviceToDevice, stream);
}
